// Attention_2010044695309
// MI455X (gfx1250) — matrix operand order repaired and checked by reading, not compiled, not run on hardware
//
#include <hip/hip_runtime.h>
#include <cmath>

#define BB 64
#define SS 4096
#define EE 128
#define DD 128
#define NEG_INF_F (-10000000000.0f)

typedef __attribute__((ext_vector_type(16))) __bf16 v16bf;
typedef __attribute__((ext_vector_type(8)))  float  v8f;
typedef __attribute__((ext_vector_type(4)))  float  v4f;

// Fast tanh: 1 - 2/(1+exp(2x)) using v_exp_f32 + v_rcp_f32 (~1e-6 rel err).
__device__ __forceinline__ float tanh_fast(float x) {
    float e = __expf(2.0f * x);
    return 1.0f - 2.0f * __builtin_amdgcn_rcpf(1.0f + e);
}

// --- Prep 1: h_proj[b][d] = hidden[b,:] . Wh[d,:] + b_attn[d]  (64x128, tiny)
__global__ void hproj_kernel(const float* __restrict__ hidden,
                             const float* __restrict__ W_attn,
                             const float* __restrict__ b_attn,
                             float* __restrict__ h_proj) {
    int b = blockIdx.x;
    int d = threadIdx.x;
    const float* h = hidden + b * DD;
    const float* w = W_attn + (size_t)d * (EE + DD);   // row d, cols [0,128) = Wh
    float acc = b_attn[d];
    #pragma unroll 8
    for (int e = 0; e < DD; ++e) acc = fmaf(h[e], w[e], acc);
    h_proj[b * DD + d] = acc;
}

// --- Prep 2: split We (cols [128,256) of W_attn) into bf16 hi/lo, row-major [d][e]
__global__ void wesplit_kernel(const float* __restrict__ W_attn,
                               __bf16* __restrict__ we_hi,
                               __bf16* __restrict__ we_lo) {
    int idx = blockIdx.x * blockDim.x + threadIdx.x;   // 0..16383
    int d = idx / EE, e = idx % EE;
    float x = W_attn[(size_t)d * (EE + DD) + DD + e];
    __bf16 h = (__bf16)x;
    __bf16 l = (__bf16)(x - (float)h);
    we_hi[idx] = h;
    we_lo[idx] = l;
}

// --- Main: fused e_proj GEMM (split-bf16 WMMA) + tanh + v-dot -> logits
// grid = (S/128, B), block = 256 (8 waves); wave w handles s-rows [s0, s0+16)
__global__ __launch_bounds__(256)
void attn_logits_kernel(const float* __restrict__ seq,      // (S, B, E)
                        const float* __restrict__ h_proj,   // (B, D)
                        const __bf16* __restrict__ we_hi,   // (D, E)
                        const __bf16* __restrict__ we_lo,   // (D, E)
                        const float* __restrict__ v_w,      // (D)
                        float* __restrict__ logits) {       // (B, S)
    const int b    = blockIdx.y;
    const int wave = threadIdx.x >> 5;
    const int lane = threadIdx.x & 31;
    const int s0   = blockIdx.x * 128 + wave * 16;
    const int row  = lane & 15;                 // A-matrix M = lane % 16
    const int abase = (lane < 16) ? 0 : 16;     // A K-pattern: {base..base+15}
    const int kb    = (lane < 16) ? 0 : 16;     // B K-half per lane group
    const int ncol  = lane & 15;                // B/C/D column N = lane % 16

    const float* rowp = seq + ((size_t)(s0 + row) * BB + b) * EE;  // contiguous 512B row
    __builtin_prefetch(rowp, 0, 0);             // global_prefetch_b8 (rows are 32KB-strided)
    __builtin_prefetch(rowp + 64, 0, 0);

    // Build A fragments (16x32 bf16 per K-chunk), split hi/lo on the fly.
    v16bf a_hi[4], a_lo[4];
    #pragma unroll
    for (int c = 0; c < 4; ++c) {
        const float* p = rowp + 32 * c + abase;
        v4f f0 = *(const v4f*)(p);
        v4f f1 = *(const v4f*)(p + 4);
        v4f f2 = *(const v4f*)(p + 8);
        v4f f3 = *(const v4f*)(p + 12);
        float xv[16];
        #pragma unroll
        for (int i = 0; i < 4; ++i) {
            xv[i]      = f0[i];
            xv[4 + i]  = f1[i];
            xv[8 + i]  = f2[i];
            xv[12 + i] = f3[i];
        }
        #pragma unroll
        for (int i = 0; i < 16; ++i) {
            __bf16 h = (__bf16)xv[i];
            a_hi[c][i] = h;
            a_lo[c][i] = (__bf16)(xv[i] - (float)h);
        }
    }

    float partial[8];
    #pragma unroll
    for (int j = 0; j < 8; ++j) partial[j] = 0.0f;

    #pragma unroll
    for (int t = 0; t < 8; ++t) {               // 8 d-tiles of 16 columns
        // Two independent accumulators break the WMMA RAW chain:
        // acc0 <- hi*hi terms, acc1 <- hi*lo + lo*hi correction terms.
        v8f acc0 = {};
        v8f acc1 = {};
        #pragma unroll
        for (int c = 0; c < 4; ++c) {
            // B fragment: lane holds column n (contiguous in e), K-half kb.
            size_t off = (size_t)(t * 16 + ncol) * EE + 32 * c + kb;
            v16bf bh = *(const v16bf*)(we_hi + off);
            v16bf bl = *(const v16bf*)(we_lo + off);
            acc0 = __builtin_amdgcn_wmma_f32_16x16x32_bf16(false, a_hi[c], false, bh,
                                                           (short)0, acc0, false, false);
            acc1 = __builtin_amdgcn_wmma_f32_16x16x32_bf16(false, a_hi[c], false, bl,
                                                           (short)0, acc1, false, false);
            acc1 = __builtin_amdgcn_wmma_f32_16x16x32_bf16(false, a_lo[c], false, bh,
                                                           (short)0, acc1, false, false);
        }
        // Epilogue: C/D layout => this lane's elements are column n, rows j+8*(lane/16)
        int n = t * 16 + ncol;
        float hp = h_proj[b * DD + n];
        float vv = v_w[n];
        #pragma unroll
        for (int j = 0; j < 8; ++j)
            partial[j] = fmaf(vv, tanh_fast((acc0[j] + acc1[j]) + hp), partial[j]);
    }

    // Reduce the 16 lanes of each half (columns) -> per-row logits
    #pragma unroll
    for (int j = 0; j < 8; ++j) {
        float p = partial[j];
        #pragma unroll
        for (int off = 1; off < 16; off <<= 1) p += __shfl_xor(p, off, 32);
        partial[j] = p;
    }
    if ((lane & 15) == 0) {
        int rbase = (lane < 16) ? 0 : 8;        // lane 0 -> rows 0..7, lane 16 -> rows 8..15
        #pragma unroll
        for (int j = 0; j < 8; ++j)
            logits[(size_t)b * SS + s0 + rbase + j] = partial[j];
    }
}

// --- Softmax over S (4096) per batch row; mask applied here. One block per b.
__global__ __launch_bounds__(256)
void softmax_kernel(const int* __restrict__ mask, float* __restrict__ out) {
    __shared__ float red[256];
    const int b = blockIdx.x, tid = threadIdx.x;
    float x[16];
    float mx = -3.4e38f;
    #pragma unroll
    for (int i = 0; i < 16; ++i) {
        int s = tid + i * 256;
        float v = out[(size_t)b * SS + s];
        v = (mask[(size_t)b * SS + s] == 0) ? NEG_INF_F : v;
        x[i] = v;
        mx = fmaxf(mx, v);
    }
    red[tid] = mx;
    __syncthreads();
    for (int off = 128; off > 0; off >>= 1) {
        if (tid < off) red[tid] = fmaxf(red[tid], red[tid + off]);
        __syncthreads();
    }
    mx = red[0];
    __syncthreads();
    float sum = 0.0f;
    #pragma unroll
    for (int i = 0; i < 16; ++i) {
        x[i] = __expf(x[i] - mx);
        sum += x[i];
    }
    red[tid] = sum;
    __syncthreads();
    for (int off = 128; off > 0; off >>= 1) {
        if (tid < off) red[tid] += red[tid + off];
        __syncthreads();
    }
    float inv = __builtin_amdgcn_rcpf(red[0]);
    #pragma unroll
    for (int i = 0; i < 16; ++i)
        out[(size_t)b * SS + tid + i * 256] = x[i] * inv;
}

extern "C" void kernel_launch(void* const* d_in, const int* in_sizes, int n_in,
                              void* d_out, int out_size, void* d_ws, size_t ws_size,
                              hipStream_t stream) {
    const float* hidden = (const float*)d_in[0];   // (64,128)
    const float* seq    = (const float*)d_in[1];   // (4096,64,128)
    const int*   mask   = (const int*)d_in[2];     // (64,4096)
    const float* W_attn = (const float*)d_in[3];   // (128,256)
    const float* b_attn = (const float*)d_in[4];   // (128)
    const float* v_w    = (const float*)d_in[5];   // (1,128)
    float* out = (float*)d_out;                    // (64,4096)

    char* ws = (char*)d_ws;
    float*  h_proj = (float*)ws;                   // 64*128*4   = 32 KB
    __bf16* we_hi  = (__bf16*)(ws + 32768);        // 128*128*2  = 32 KB
    __bf16* we_lo  = (__bf16*)(ws + 65536);        // 128*128*2  = 32 KB

    hproj_kernel<<<dim3(BB), dim3(DD), 0, stream>>>(hidden, W_attn, b_attn, h_proj);
    wesplit_kernel<<<dim3(64), dim3(256), 0, stream>>>(W_attn, we_hi, we_lo);
    attn_logits_kernel<<<dim3(SS / 128, BB), dim3(256), 0, stream>>>(
        seq, h_proj, we_hi, we_lo, v_w, out);
    softmax_kernel<<<dim3(BB), dim3(256), 0, stream>>>(mask, out);
}
